// AttentionalFactorizationMachine_86517821214538
// MI455X (gfx1250) — matrix operand order repaired and checked by reading, not compiled, not run on hardware
//
#include <hip/hip_runtime.h>
#include <math.h>

#define NBATCH 2048
#define NF 40
#define ED 64
#define AS 32
#define NPAIR 780   // 40*39/2
#define NPAD 784    // 49 tiles of 16
#define NTILES 49

typedef __attribute__((ext_vector_type(16))) _Float16 v16h;
typedef __attribute__((ext_vector_type(8)))  float    v8f;

union V16H { v16h v; _Float16 h[16]; };

__global__ __launch_bounds__(256)
void afm_wmma_kernel(const float* __restrict__ x,
                     const float* __restrict__ attn_w,
                     const float* __restrict__ attn_b,
                     const float* __restrict__ proj_w,
                     const float* __restrict__ fc_w,
                     const float* __restrict__ fc_b,
                     float* __restrict__ out)
{
    __shared__ float      xs[NF * ED];        // x[b] staged (10 KB)
    __shared__ _Float16   bwh[ED * AS];       // attn_w in f16 (4 KB)
    __shared__ float      L[NPAD];            // logits -> softmax scores
    __shared__ float      red[256];           // reductions
    __shared__ float      part[4 * 64];       // weighted-sum partials
    __shared__ unsigned char rowA[NPAIR];
    __shared__ unsigned char colA[NPAIR];

    const int t    = threadIdx.x;
    const int b    = blockIdx.x;
    const int lane = t & 31;
    const int wave = t >> 5;
    const int lg   = lane >> 4;   // lane group (0/1)
    const int ll   = lane & 15;

    // ---------------- Phase 0: stage inputs into LDS ----------------
    const float4* xg  = (const float4*)(x + (size_t)b * NF * ED);
    float4*       xs4 = (float4*)xs;
    for (int i = t; i < NF * ED / 4; i += 256) xs4[i] = xg[i];
    for (int i = t; i < ED * AS;     i += 256) bwh[i] = (_Float16)attn_w[i];
    for (int i = t; i < NPAD;        i += 256) L[i] = 0.0f;
    if (t < NF) {
        int r = t;
        int off = r * (NF - 1) - (r * (r - 1)) / 2;
        for (int c = r + 1; c < NF; ++c) {
            rowA[off] = (unsigned char)r;
            colA[off] = (unsigned char)c;
            ++off;
        }
    }
    __syncthreads();

    // ---------------- Phase 1: h = relu(IP @ Wa + ba); logits = h @ pw ----
    // B operand (32x16 f16): lane covers n = nt*16 + (lane&15),
// slot e covers k = kt*32 + (e&7) + 8*(lane>>4) + 16*(e>>3)
    V16H b00, b01, b10, b11;
#pragma unroll
    for (int e = 0; e < 16; ++e) {
        int k0 = (e & 7) + 8 * lg + 16 * (e >> 3);
        b00.h[e] = bwh[(k0     ) * AS + ll     ];
        b01.h[e] = bwh[(k0     ) * AS + ll + 16];
        b10.h[e] = bwh[(k0 + 32) * AS + ll     ];
        b11.h[e] = bwh[(k0 + 32) * AS + ll + 16];
    }
    const float ba0 = attn_b[ll], ba1 = attn_b[ll + 16];
    const float pw0 = proj_w[ll], pw1 = proj_w[ll + 16];

    for (int mt = wave; mt < NTILES; mt += 8) {     // uniform per wave
        const int  m     = mt * 16 + ll;            // A row handled by this lane
        const bool valid = (m < NPAIR);
        const int  r = valid ? (int)rowA[m] : 0;
        const int  c = valid ? (int)colA[m] : 0;
        const float* xr = xs + r * ED;
        const float* xc = xs + c * ED;

        // A operand (16x32 f16): row m = mt*16+(lane&15),
        // slot e covers k = kt*32 + (e&7) + 8*(lane>>4) + 16*(e>>3)
        V16H a0, a1;
#pragma unroll
        for (int e = 0; e < 16; ++e) {
            int k = (e & 7) + 8 * lg + 16 * (e >> 3);
            a0.h[e] = valid ? (_Float16)(xr[k     ] * xc[k     ]) : (_Float16)0.0f;
            a1.h[e] = valid ? (_Float16)(xr[k + 32] * xc[k + 32]) : (_Float16)0.0f;
        }

        v8f c0 = {}; v8f c1 = {};
        c0 = __builtin_amdgcn_wmma_f32_16x16x32_f16(false, a0.v, false, b00.v, (short)0, c0, false, false);
        c0 = __builtin_amdgcn_wmma_f32_16x16x32_f16(false, a1.v, false, b10.v, (short)0, c0, false, false);
        c1 = __builtin_amdgcn_wmma_f32_16x16x32_f16(false, a0.v, false, b01.v, (short)0, c1, false, false);
        c1 = __builtin_amdgcn_wmma_f32_16x16x32_f16(false, a1.v, false, b11.v, (short)0, c1, false, false);

        // C layout: lane holds column n = ll (+16 for tile 1), rows m = j + 8*lg.
        // Fuse bias + relu + proj_w, reduce over n via LDS float atomics.
#pragma unroll
        for (int j = 0; j < 8; ++j) {
            float h0 = fmaxf(c0[j] + ba0, 0.0f);
            float h1 = fmaxf(c1[j] + ba1, 0.0f);
            atomicAdd(&L[mt * 16 + j + 8 * lg], h0 * pw0 + h1 * pw1);
        }
    }
    __syncthreads();

    // ---------------- Phase 2: softmax over 780 logits (proj_b invariant) --
    float lm = -1e30f;
    for (int p = t; p < NPAIR; p += 256) lm = fmaxf(lm, L[p]);
    red[t] = lm; __syncthreads();
    for (int s = 128; s > 0; s >>= 1) { if (t < s) red[t] = fmaxf(red[t], red[t + s]); __syncthreads(); }
    const float mx = red[0];
    __syncthreads();

    float ls = 0.0f;
    for (int p = t; p < NPAIR; p += 256) { float e = __expf(L[p] - mx); L[p] = e; ls += e; }
    red[t] = ls; __syncthreads();
    for (int s = 128; s > 0; s >>= 1) { if (t < s) red[t] += red[t + s]; __syncthreads(); }
    const float inv = 1.0f / red[0];
    __syncthreads();
    for (int p = t; p < NPAIR; p += 256) L[p] *= inv;
    __syncthreads();

    // ---------------- Phase 3: attn_out[d] = sum_p score[p]*x[r][d]*x[c][d]
    const int d = t & 63;
    const int g = t >> 6;
    float acc = 0.0f;
    for (int p = g; p < NPAIR; p += 4)
        acc += L[p] * xs[(int)rowA[p] * ED + d] * xs[(int)colA[p] * ED + d];
    part[g * 64 + d] = acc;
    __syncthreads();

    // ---------------- Phase 4: out[b] = attn_out . fc_w + fc_b -------------
    float v = 0.0f;
    if (t < 64) {
        float ao = part[t] + part[64 + t] + part[128 + t] + part[192 + t];
        v = ao * fc_w[t];
    }
    red[t] = v; __syncthreads();
    for (int s = 128; s > 0; s >>= 1) { if (t < s) red[t] += red[t + s]; __syncthreads(); }
    if (t == 0) out[b] = red[0] + fc_b[0];
}

extern "C" void kernel_launch(void* const* d_in, const int* in_sizes, int n_in,
                              void* d_out, int out_size, void* d_ws, size_t ws_size,
                              hipStream_t stream) {
    const float* x      = (const float*)d_in[0];
    const float* attn_w = (const float*)d_in[1];
    const float* attn_b = (const float*)d_in[2];
    const float* proj_w = (const float*)d_in[3];
    // d_in[4] = proj_b: constant shift of all logits -> softmax-invariant, unused
    const float* fc_w   = (const float*)d_in[5];
    const float* fc_b   = (const float*)d_in[6];
    afm_wmma_kernel<<<NBATCH, 256, 0, stream>>>(x, attn_w, attn_b, proj_w,
                                                fc_w, fc_b, (float*)d_out);
}
